// CARAFE_downsample_30880814858247
// MI455X (gfx1250) — matrix operand order repaired and checked by reading, not compiled, not run on hardware
//
#include <hip/hip_runtime.h>
#include <hip/hip_bf16.h>
#include <math.h>

// ---------------------------------------------------------------------------
// CARAFE-downsample for MI455X (gfx1250, wave32, WMMA)
//   k1: 1x1 conv (fp32 WMMA 16x16x4)  -> y bf16, channel-last [b][h][w][32]
//   k2: 5x5/s2 conv (bf16 WMMA 16x16x32, 25 taps = 25 K-steps) + bias+softmax
//   k3: weighted 5x5 reassembly (memory-bound VALU)
// ---------------------------------------------------------------------------

typedef float  v2f   __attribute__((ext_vector_type(2)));
typedef float  v8f   __attribute__((ext_vector_type(8)));
typedef __bf16 v8bf  __attribute__((ext_vector_type(8)));
typedef __bf16 v16bf __attribute__((ext_vector_type(16)));

#define B_    4
#define CIN   64
#define CMID  32
#define H_    256
#define W_    256
#define HW_   (H_ * W_)
#define HO_   128
#define WO_   128
#define KUP   5
#define K2    25
#define K2P   32   // outch padded to 32 for two 16-row WMMA tiles

__device__ __forceinline__ v16bf zero16() {
  v16bf v;
#pragma unroll
  for (int i = 0; i < 16; ++i) v[i] = (__bf16)0.0f;
  return v;
}

// ---------------------------------------------------------------------------
// Kernel 0: repack enc_w (25,32,5,5) f32 -> bf16 [tap(25)][outch(32,pad0)][c(32)]
// ---------------------------------------------------------------------------
__global__ void carafe_prep_encw(const float* __restrict__ enc_w,
                                 __bf16* __restrict__ encw_bf) {
  int i = blockIdx.x * blockDim.x + threadIdx.x;
  if (i >= K2 * K2P * CMID) return;
  int c = i & 31;
  int m = (i >> 5) & 31;
  int t = i >> 10;
  float v = 0.0f;
  if (m < K2) v = enc_w[(m * CMID + c) * K2 + t];  // ((m*32+c)*5+kh)*5+kw, t=kh*5+kw
  encw_bf[i] = (__bf16)v;
}

// ---------------------------------------------------------------------------
// Kernel 1: 1x1 channel compression via V_WMMA_F32_16X16X4_F32
//   GEMM: y[32, pix] = comp_w[32,64] * x[64, pix] + comp_b
//   wave = 32 outch x 16 pixels; K=64 -> 16 steps x 2 outch tiles
//   output: bf16 channel-last  ybf[((b*HW)+pix)*32 + ch]
// ---------------------------------------------------------------------------
__global__ void carafe_compress(const float* __restrict__ x,
                                const float* __restrict__ cw,
                                const float* __restrict__ cb,
                                __bf16* __restrict__ ybf) {
  int lane = threadIdx.x & 31;
  int wave = blockIdx.x * (blockDim.x >> 5) + (threadIdx.x >> 5);
  int h1  = lane >> 4;    // lane half (0/1)
  int col = lane & 15;    // pixel column within 16-wide strip / outch row
  int pix0 = wave << 4;                 // strip base in flattened b*HW
  int b = pix0 >> 16;                   // 65536 pixels per image
  int p = (pix0 & (HW_ - 1)) + col;     // per-lane pixel within image

  const float* xb = x + (size_t)b * CIN * HW_ + p;
  v8f acc0 = {};
  v8f acc1 = {};
#pragma unroll
  for (int kb = 0; kb < CIN; kb += 4) {
    // B (4x16 f32): vgpr0 = row(h1), vgpr1 = row(2+h1), col = lane&15
    v2f bf;
    bf.x = xb[(size_t)(kb + h1) * HW_];
    bf.y = xb[(size_t)(kb + 2 + h1) * HW_];
    // A (16x4 f32): lane holds M=col, K = kb+2*h1 .. +1 (consecutive -> v2f load)
    v2f a0 = *(const v2f*)(cw + (size_t)col * CIN + kb + 2 * h1);
    v2f a1 = *(const v2f*)(cw + (size_t)(col + 16) * CIN + kb + 2 * h1);
    acc0 = __builtin_amdgcn_wmma_f32_16x16x4_f32(false, a0, false, bf, (short)0, acc0, false, false);
    acc1 = __builtin_amdgcn_wmma_f32_16x16x4_f32(false, a1, false, bf, (short)0, acc1, false, false);
  }
  // Epilogue: bias + bf16 convert + channel-last store.
  // C/D layout: vgpr r, lane l -> (M = r + 8*h1, N = col) -> lane owns 8
  // consecutive channels starting at 8*h1 (acc0) / 16+8*h1 (acc1).
  __bf16* yp = ybf + ((size_t)b * HW_ + p) * CMID;
  v8bf o0, o1;
#pragma unroll
  for (int r = 0; r < 8; ++r) {
    o0[r] = (__bf16)(acc0[r] + cb[8 * h1 + r]);
    o1[r] = (__bf16)(acc1[r] + cb[16 + 8 * h1 + r]);
  }
  *(v8bf*)(yp + 8 * h1)      = o0;   // 16B aligned store
  *(v8bf*)(yp + 16 + 8 * h1) = o1;
}

// ---------------------------------------------------------------------------
// Kernel 2: 5x5 stride-2 content encoder via V_WMMA_F32_16X16X32_BF16
//   25 taps, each tap is one K=32 (channels) WMMA per 16-outch tile.
//   Fused bias + in-register softmax over 25 channels (lane-pair shfl_xor 16).
//   wave = one (b, ho) row, 16 consecutive wo.
//   output: wout[((b*HO+ho)*WO+wo)*32 + ch]  (pixel-major for k3)
// ---------------------------------------------------------------------------
__global__ void carafe_encoder_softmax(const __bf16* __restrict__ ybf,
                                       const __bf16* __restrict__ encw,
                                       const float* __restrict__ eb,
                                       float* __restrict__ wout) {
  int lane = threadIdx.x & 31;
  int wave = blockIdx.x * (blockDim.x >> 5) + (threadIdx.x >> 5);
  int h1  = lane >> 4;
  int col = lane & 15;
  int b   = wave >> 10;            // 128 rows * 8 strips = 1024 per image
  int rem = wave & 1023;
  int ho  = rem >> 3;
  int wo  = ((rem & 7) << 4) + col;
  int win_base = 2 * wo - 2;

  v8f acc0 = {};
  v8f acc1 = {};
  for (int kh = 0; kh < KUP; ++kh) {
    int hin = 2 * ho - 2 + kh;
    if (hin < 0 || hin >= H_) continue;            // uniform across wave
    const __bf16* yrow = ybf + ((size_t)(b * H_ + hin) * W_) * CMID;
#pragma unroll
    for (int kw = 0; kw < KUP; ++kw) {
      int win = win_base + kw;
      // B (32x16 bf16): lane holds col=wo, K=channels 16*h1 .. +15 -> one
      // aligned 32B load from channel-last y.
      v16bf bfrag = zero16();
      if (win >= 0 && win < W_)
        bfrag = *(const v16bf*)(yrow + (size_t)win * CMID + 16 * h1);
      int t = kh * KUP + kw;
      // A (16x32 bf16): lane M=col(+16), K = 16h..16h+15 (contiguous)
      const v8bf* w0 = (const v8bf*)(encw + (size_t)t * (K2P * CMID) + (size_t)col * CMID);
      const v8bf* w1 = (const v8bf*)(encw + (size_t)t * (K2P * CMID) + (size_t)(col + 16) * CMID);
      v16bf a0 = __builtin_shufflevector(w0[2 * h1], w0[2 * h1 + 1],
                                         0,1,2,3,4,5,6,7,8,9,10,11,12,13,14,15);
      v16bf a1 = __builtin_shufflevector(w1[2 * h1], w1[2 * h1 + 1],
                                         0,1,2,3,4,5,6,7,8,9,10,11,12,13,14,15);
      acc0 = __builtin_amdgcn_wmma_f32_16x16x32_bf16(false, a0, false, bfrag, (short)0, acc0, false, false);
      acc1 = __builtin_amdgcn_wmma_f32_16x16x32_bf16(false, a1, false, bfrag, (short)0, acc1, false, false);
    }
  }

  // Bias + softmax over 25 channels. Lane l holds channels {8h..8h+7} and
  // {16+8h..16+8h+7}; pad channels (>=25) forced to -inf.
  float v[16];
#pragma unroll
  for (int r = 0; r < 8; ++r) v[r] = acc0[r] + eb[8 * h1 + r];
#pragma unroll
  for (int r = 0; r < 8; ++r) {
    int ch = 16 + 8 * h1 + r;
    v[8 + r] = (ch < K2) ? (acc1[r] + eb[ch]) : -INFINITY;
  }
  float m = v[0];
#pragma unroll
  for (int i = 1; i < 16; ++i) m = fmaxf(m, v[i]);
  m = fmaxf(m, __shfl_xor(m, 16, 32));   // combine lane pair (l, l^16)
  float e[16];
  float s = 0.0f;
#pragma unroll
  for (int i = 0; i < 16; ++i) { e[i] = __expf(v[i] - m); s += e[i]; }
  s += __shfl_xor(s, 16, 32);
  float inv = 1.0f / s;

  float* op = wout + ((size_t)(b * HO_ + ho) * WO_ + wo) * K2P;
#pragma unroll
  for (int r = 0; r < 8; ++r) op[8 * h1 + r] = e[r] * inv;
#pragma unroll
  for (int r = 0; r < 8; ++r) op[16 + 8 * h1 + r] = e[8 + r] * inv;
}

// ---------------------------------------------------------------------------
// Kernel 3: weighted reassembly  out[b,c,ho,wo] = sum_t x[b,c,2ho-2+kh,2wo-2+kw]*w[t]
//   thread = (b,ho,wo); 25 weights in registers, loop c. wo-major => coalesced.
// ---------------------------------------------------------------------------
__global__ void carafe_reassemble(const float* __restrict__ x,
                                  const float* __restrict__ wbuf,
                                  float* __restrict__ out) {
  int idx = blockIdx.x * blockDim.x + threadIdx.x;   // B*HO*WO = 65536
  int b   = idx >> 14;
  int rem = idx & 16383;
  int ho  = rem >> 7;
  int wo  = rem & 127;

  float wk[K2];
  const float* wp = wbuf + ((size_t)(b * HO_ + ho) * WO_ + wo) * K2P;
#pragma unroll
  for (int t = 0; t < K2; ++t) wk[t] = wp[t];

  int hbase = 2 * ho - 2, wbase = 2 * wo - 2;
  int kh0 = hbase < 0 ? -hbase : 0;
  int kh1 = (hbase + KUP > H_) ? (H_ - hbase) : KUP;
  int kw0 = wbase < 0 ? -wbase : 0;
  int kw1 = (wbase + KUP > W_) ? (W_ - wbase) : KUP;

  const float* xb = x + (size_t)b * CIN * HW_;
  float* ob = out + (size_t)b * CIN * (HO_ * WO_) + (size_t)ho * WO_ + wo;
  for (int c = 0; c < CIN; ++c) {
    const float* xc = xb + (size_t)c * HW_;
    float s = 0.0f;
    for (int kh = kh0; kh < kh1; ++kh) {
      const float* row = xc + (size_t)(hbase + kh) * W_ + wbase;
      const float* wr  = wk + kh * KUP;
      for (int kw = kw0; kw < kw1; ++kw) s = fmaf(row[kw], wr[kw], s);
    }
    ob[(size_t)c * (HO_ * WO_)] = s;
  }
}

// ---------------------------------------------------------------------------
// Launch. Workspace layout (all 256B-aligned):
//   [0, 16 MiB)            y bf16   : 4*256*256*32 * 2B = 16,777,216 B
//   [16 MiB, +64 KiB)      enc_w bf16 repack: 25*32*32 * 2B = 51,200 B
//   [16 MiB + 64 KiB, ...) softmax weights f32: 4*128*128*32 * 4B = 8,388,608 B
// Total ~25.2 MB.
// ---------------------------------------------------------------------------
extern "C" void kernel_launch(void* const* d_in, const int* in_sizes, int n_in,
                              void* d_out, int out_size, void* d_ws, size_t ws_size,
                              hipStream_t stream) {
  const float* x      = (const float*)d_in[0];
  const float* comp_w = (const float*)d_in[1];
  const float* comp_b = (const float*)d_in[2];
  const float* enc_w  = (const float*)d_in[3];
  const float* enc_b  = (const float*)d_in[4];
  float* out = (float*)d_out;

  char* ws = (char*)d_ws;
  __bf16* ybf  = (__bf16*)ws;
  __bf16* encw = (__bf16*)(ws + (size_t)(16u << 20));
  float*  wbuf = (float*)(ws + (size_t)(16u << 20) + (size_t)(64u << 10));

  // k0: weight repack (25600 elems)
  carafe_prep_encw<<<100, 256, 0, stream>>>(enc_w, encw);
  // k1: 16384 strips of 16 pixels, 8 waves/block
  carafe_compress<<<2048, 256, 0, stream>>>(x, comp_w, comp_b, ybf);
  // k2: 4096 strips (b * 128 rows * 8), 8 waves/block
  carafe_encoder_softmax<<<512, 256, 0, stream>>>(ybf, encw, enc_b, wbuf);
  // k3: 65536 output pixels
  carafe_reassemble<<<256, 256, 0, stream>>>(x, wbuf, out);
}
